// HeteroGNN_with_RoBERTa_13417477833281
// MI455X (gfx1250) — matrix operand order repaired and checked by reading, not compiled, not run on hardware
//
#include <hip/hip_runtime.h>
#include <hip/hip_bf16.h>

typedef __bf16 bf16;
typedef __attribute__((ext_vector_type(16))) __bf16 v16bf;
typedef __attribute__((ext_vector_type(8)))  __bf16 v8bf;
typedef __attribute__((ext_vector_type(8)))  float  v8f;

#define NE   50000
#define NM   30000
#define BB   16
#define DIN  300
#define C_   768
#define H1   4
#define HC1  3072
#define EDIM 3072
#define AH   12
#define HD   256

// ---------------------------------------------------------------------------
// Weight transpose + fp32->bf16 convert:  W[K,N] -> Wt[N,Kp]  (Kp mult of 32,
// zero padded) so GEMM B-tile loads are always aligned & unguarded.
// ---------------------------------------------------------------------------
__global__ void wconv_t(const float* __restrict__ W, bf16* __restrict__ Wt,
                        int K, int N, int Kp) {
  __shared__ float tile[32][33];
  int n0 = blockIdx.x * 32, k0 = blockIdx.y * 32;
  int tx = threadIdx.x, ty = threadIdx.y;       // 32 x 8
  #pragma unroll
  for (int i = 0; i < 4; ++i) {
    int k = k0 + ty + i * 8, n = n0 + tx;
    tile[ty + i * 8][tx] = (k < K && n < N) ? W[(size_t)k * N + n] : 0.0f;
  }
  __syncthreads();
  #pragma unroll
  for (int i = 0; i < 4; ++i) {
    int n = n0 + ty + i * 8, kk = k0 + tx;
    if (n < N && kk < Kp) Wt[(size_t)n * Kp + kk] = (bf16)tile[tx][ty + i * 8];
  }
}

// ---------------------------------------------------------------------------
// W-tilde: wred[d,h] = sum_c W[d, h*C+c]*att[h*C+c]   (a = X @ wred trick)
// ---------------------------------------------------------------------------
__global__ void redw_kernel(const float* __restrict__ W, const float* __restrict__ att,
                            float* __restrict__ out, int K, int H, int C) {
  int idx = blockIdx.x * blockDim.x + threadIdx.x;
  if (idx >= K * H) return;
  int d = idx / H, h = idx - d * H;
  const float* wrow = W + (size_t)d * (H * C) + h * C;
  const float* arow = att + h * C;
  float s = 0.f;
  for (int c = 0; c < C; ++c) s += wrow[c] * arow[c];
  out[d * H + h] = s;
}

// out[row,h] = X[row,:] @ Wr[:,h]   (block of 128 per row, LDS reduce)
template <int H>
__global__ void matvec_h(const float* __restrict__ X, const float* __restrict__ Wr,
                         float* __restrict__ out, int K) {
  __shared__ float red[128 * H];
  int row = blockIdx.x, t = threadIdx.x;
  float acc[H];
  #pragma unroll
  for (int h = 0; h < H; ++h) acc[h] = 0.f;
  const float* xr = X + (size_t)row * K;
  for (int c = t; c < K; c += 128) {
    float x = xr[c];
    #pragma unroll
    for (int h = 0; h < H; ++h) acc[h] += x * Wr[c * H + h];
  }
  #pragma unroll
  for (int h = 0; h < H; ++h) red[t * H + h] = acc[h];
  __syncthreads();
  for (int off = 64; off > 0; off >>= 1) {
    if (t < off) {
      #pragma unroll
      for (int h = 0; h < H; ++h) red[t * H + h] += red[(t + off) * H + h];
    }
    __syncthreads();
  }
  if (t < H) out[row * H + t] = red[t];
}

// ---------------------------------------------------------------------------
// CSR build: histogram -> single-block scan -> bucket fill
// ---------------------------------------------------------------------------
__global__ void hist_kernel(const int* __restrict__ dstc, int E, int* cnt) {
  int i = blockIdx.x * blockDim.x + threadIdx.x;
  if (i < E) atomicAdd(&cnt[dstc[i]], 1);
}
__global__ void scan_kernel(const int* __restrict__ cnt, int* __restrict__ offs, int n) {
  __shared__ int buf[1024];
  __shared__ int carry;
  int t = threadIdx.x;
  if (t == 0) { carry = 0; offs[0] = 0; }
  __syncthreads();
  for (int base = 0; base < n; base += 1024) {
    int i = base + t;
    buf[t] = (i < n) ? cnt[i] : 0;
    __syncthreads();
    for (int off = 1; off < 1024; off <<= 1) {
      int x = (t >= off) ? buf[t - off] : 0;
      __syncthreads();
      buf[t] += x;
      __syncthreads();
    }
    if (i < n) offs[i + 1] = carry + buf[t];
    __syncthreads();
    if (t == 0) carry += buf[1023];
    __syncthreads();
  }
}
__global__ void copy_i32(const int* a, int* b, int n) {
  int i = blockIdx.x * blockDim.x + threadIdx.x;
  if (i < n) b[i] = a[i];
}
__global__ void fill_kernel(const int* __restrict__ dstc, int E, int* cursor, int* eidx) {
  int i = blockIdx.x * blockDim.x + threadIdx.x;
  if (i < E) { int p = atomicAdd(&cursor[dstc[i]], 1); eidx[p] = i; }
}

// ---------------------------------------------------------------------------
// Per-dst segment max + sum(exp) of leaky_relu(a_s[src]+a_d[dst])
// ---------------------------------------------------------------------------
template <int H>
__global__ void gat_stats(const int* __restrict__ offs, const int* __restrict__ eidx,
                          const int* __restrict__ srcc,
                          const float* __restrict__ as_, const float* __restrict__ ad_,
                          float* __restrict__ m_out, float* __restrict__ den_out) {
  __shared__ float red[128 * H];
  int d = blockIdx.x, t = threadIdx.x;
  int s0 = offs[d], s1 = offs[d + 1];
  float adl[H], mh[H];
  #pragma unroll
  for (int h = 0; h < H; ++h) { adl[h] = ad_[d * H + h]; mh[h] = -INFINITY; }
  for (int i = s0 + t; i < s1; i += 128) {
    int s = srcc[eidx[i]];
    #pragma unroll
    for (int h = 0; h < H; ++h) {
      float v = as_[s * H + h] + adl[h];
      v = v > 0.f ? v : 0.2f * v;
      mh[h] = fmaxf(mh[h], v);
    }
  }
  #pragma unroll
  for (int h = 0; h < H; ++h) red[t * H + h] = mh[h];
  __syncthreads();
  for (int off = 64; off > 0; off >>= 1) {
    if (t < off) {
      #pragma unroll
      for (int h = 0; h < H; ++h)
        red[t * H + h] = fmaxf(red[t * H + h], red[(t + off) * H + h]);
    }
    __syncthreads();
  }
  float mfin[H];
  #pragma unroll
  for (int h = 0; h < H; ++h) mfin[h] = red[h];
  __syncthreads();
  float sh[H];
  #pragma unroll
  for (int h = 0; h < H; ++h) sh[h] = 0.f;
  for (int i = s0 + t; i < s1; i += 128) {
    int s = srcc[eidx[i]];
    #pragma unroll
    for (int h = 0; h < H; ++h) {
      float v = as_[s * H + h] + adl[h];
      v = v > 0.f ? v : 0.2f * v;
      sh[h] += __expf(v - mfin[h]);
    }
  }
  #pragma unroll
  for (int h = 0; h < H; ++h) red[t * H + h] = sh[h];
  __syncthreads();
  for (int off = 64; off > 0; off >>= 1) {
    if (t < off) {
      #pragma unroll
      for (int h = 0; h < H; ++h) red[t * H + h] += red[(t + off) * H + h];
    }
    __syncthreads();
  }
  if (t == 0) {
    #pragma unroll
    for (int h = 0; h < H; ++h) {
      m_out[d * H + h] = mfin[h];
      den_out[d * H + h] = red[h];
    }
  }
}

template <int H>
__global__ void gat_alpha(const int* __restrict__ srcc, const int* __restrict__ dstc, int E,
                          const float* __restrict__ as_, const float* __restrict__ ad_,
                          const float* __restrict__ m_, const float* __restrict__ den_,
                          float* __restrict__ alpha) {
  int idx = blockIdx.x * blockDim.x + threadIdx.x;
  if (idx >= E * H) return;
  int e = idx / H, h = idx - e * H;
  int s = srcc[e], d = dstc[e];
  float v = as_[s * H + h] + ad_[d * H + h];
  v = v > 0.f ? v : 0.2f * v;
  alpha[idx] = __expf(v - m_[d * H + h]) / (den_[d * H + h] + 1e-16f);
}

// ---------------------------------------------------------------------------
// Gather aggregation in INPUT space: Z[dst,h,:] = sum_e alpha[e,h]*X[src_e,:]
// (prefetch the next edge's source row: latency-bound random gathers)
// ---------------------------------------------------------------------------
template <int H, int CPT>
__global__ void gat_gather(const int* __restrict__ offs, const int* __restrict__ eidx,
                           const int* __restrict__ srcc, const float* __restrict__ alpha,
                           const float* __restrict__ X, int din, float* __restrict__ Z) {
  int d = blockIdx.x, t = threadIdx.x;
  int s0 = offs[d], s1 = offs[d + 1];
  float acc[H][CPT] = {};
  for (int i = s0; i < s1; ++i) {
    int e = eidx[i], s = srcc[e];
    if (i + 1 < s1) {
      int en = eidx[i + 1];
      __builtin_prefetch(&X[(size_t)srcc[en] * din + t], 0, 1);
    }
    float al[H];
    #pragma unroll
    for (int h = 0; h < H; ++h) al[h] = alpha[e * H + h];
    const float* xr = X + (size_t)s * din;
    #pragma unroll
    for (int j = 0; j < CPT; ++j) {
      int c = t + j * 128;
      if (c < din) {
        float x = xr[c];
        #pragma unroll
        for (int h = 0; h < H; ++h) acc[h][j] += al[h] * x;
      }
    }
  }
  #pragma unroll
  for (int j = 0; j < CPT; ++j) {
    int c = t + j * 128;
    if (c < din) {
      #pragma unroll
      for (int h = 0; h < H; ++h) Z[((size_t)d * H + h) * din + c] = acc[h][j];
    }
  }
}

// split variant for tiny n_dst (father): blockIdx.y = partition, atomic finish
template <int H, int CPT>
__global__ void gat_gather_split(const int* __restrict__ offs, const int* __restrict__ eidx,
                                 const int* __restrict__ srcc, const float* __restrict__ alpha,
                                 const float* __restrict__ X, int din, float* __restrict__ Z) {
  int d = blockIdx.x, t = threadIdx.x;
  int part = blockIdx.y, P = gridDim.y;
  int s0 = offs[d], s1 = offs[d + 1];
  float acc[H][CPT] = {};
  for (int i = s0 + part; i < s1; i += P) {
    int e = eidx[i], s = srcc[e];
    if (i + P < s1) {
      int en = eidx[i + P];
      __builtin_prefetch(&X[(size_t)srcc[en] * din + t], 0, 1);
    }
    float al[H];
    #pragma unroll
    for (int h = 0; h < H; ++h) al[h] = alpha[e * H + h];
    const float* xr = X + (size_t)s * din;
    #pragma unroll
    for (int j = 0; j < CPT; ++j) {
      int c = t + j * 128;
      if (c < din) {
        float x = xr[c];
        #pragma unroll
        for (int h = 0; h < H; ++h) acc[h][j] += al[h] * x;
      }
    }
  }
  #pragma unroll
  for (int j = 0; j < CPT; ++j) {
    int c = t + j * 128;
    if (c < din) {
      #pragma unroll
      for (int h = 0; h < H; ++h)
        atomicAdd(&Z[((size_t)d * H + h) * din + c], acc[h][j]);
    }
  }
}

// ---------------------------------------------------------------------------
// bf16 WMMA GEMM:  C[M,N] (+)= A[M,K](fp32) @ Bt[N,Kp](bf16, K-major) + bias
// block 128x128, BK=32, 8 waves (4 M x 2 N), wave tile 32x64 (2x4 wmma).
// Register double-buffering: global loads for step k+1 overlap WMMA of step k.
// N must be a multiple of 128 (true for all call sites). Kp mult of 32.
// ---------------------------------------------------------------------------
#define LDS_S 40
__global__ __launch_bounds__(256) void gemm_bf16(
    const float* __restrict__ A, int lda,
    const bf16* __restrict__ Bt, int ldb,
    float* __restrict__ C, int ldc,
    const float* __restrict__ bias,
    int M, int N, int K, int accum, int relu) {
  __shared__ bf16 sA[128 * LDS_S];
  __shared__ bf16 sB[128 * LDS_S];
  int tid = threadIdx.x;
  int lane = tid & 31, wave = tid >> 5;
  int wm = wave & 3, wn = wave >> 2;
  int m0 = blockIdx.y * 128, n0 = blockIdx.x * 128;
  int half = lane >> 4, l16 = lane & 15;
  int lr = tid >> 1, lc0 = (tid & 1) * 16;
  int grow = m0 + lr;
  const float* aRow = A + (size_t)grow * lda;   // guarded by (grow < M) at use
  const bf16* bRow = Bt + (size_t)(n0 + lr) * ldb;
  bool aValid = (grow < M);

  v8f acc[2][4] = {};
  float areg[16];
  v8bf breg0, breg1;
  int nk = (K + 31) / 32;

  // ---- prologue: load tiles for kt = 0 ----
  {
    int k0 = 0;
    breg0 = *(const v8bf*)(bRow + k0 + lc0);
    breg1 = *(const v8bf*)(bRow + k0 + lc0 + 8);
    if (aValid) {
      #pragma unroll
      for (int q = 0; q < 4; ++q) {
        int kk = k0 + lc0 + q * 4;
        if (kk + 3 < K) {
          float4 v = *(const float4*)(aRow + kk);
          areg[q * 4 + 0] = v.x; areg[q * 4 + 1] = v.y;
          areg[q * 4 + 2] = v.z; areg[q * 4 + 3] = v.w;
        } else {
          #pragma unroll
          for (int i2 = 0; i2 < 4; ++i2)
            areg[q * 4 + i2] = (kk + i2 < K) ? aRow[kk + i2] : 0.f;
        }
      }
    } else {
      #pragma unroll
      for (int i2 = 0; i2 < 16; ++i2) areg[i2] = 0.f;
    }
  }

  for (int kt = 0; kt < nk; ++kt) {
    // ---- stage current tiles into LDS ----
    {
      v8bf lo, hi;
      #pragma unroll
      for (int i2 = 0; i2 < 8; ++i2) { lo[i2] = (bf16)areg[i2]; hi[i2] = (bf16)areg[8 + i2]; }
      *(v8bf*)&sA[lr * LDS_S + lc0] = lo;
      *(v8bf*)&sA[lr * LDS_S + lc0 + 8] = hi;
      *(v8bf*)&sB[lr * LDS_S + lc0] = breg0;
      *(v8bf*)&sB[lr * LDS_S + lc0 + 8] = breg1;
    }
    __syncthreads();
    // ---- prefetch next tiles into registers (overlaps WMMA below) ----
    if (kt + 1 < nk) {
      int k0 = (kt + 1) * 32;
      breg0 = *(const v8bf*)(bRow + k0 + lc0);
      breg1 = *(const v8bf*)(bRow + k0 + lc0 + 8);
      if (aValid) {
        #pragma unroll
        for (int q = 0; q < 4; ++q) {
          int kk = k0 + lc0 + q * 4;
          if (kk + 3 < K) {
            float4 v = *(const float4*)(aRow + kk);
            areg[q * 4 + 0] = v.x; areg[q * 4 + 1] = v.y;
            areg[q * 4 + 2] = v.z; areg[q * 4 + 3] = v.w;
          } else {
            #pragma unroll
            for (int i2 = 0; i2 < 4; ++i2)
              areg[q * 4 + i2] = (kk + i2 < K) ? aRow[kk + i2] : 0.f;
          }
        }
      }
    }
    // ---- compute from LDS ----
    v16bf af[2], bfr[4];
    #pragma unroll
    for (int ms = 0; ms < 2; ++ms) {
      int r = wm * 32 + ms * 16 + l16;
      v8bf lo = *(const v8bf*)&sA[r * LDS_S + half * 8];
      v8bf hi = *(const v8bf*)&sA[r * LDS_S + 16 + half * 8];
      af[ms] = __builtin_shufflevector(lo, hi, 0,1,2,3,4,5,6,7,8,9,10,11,12,13,14,15);
    }
    #pragma unroll
    for (int ns = 0; ns < 4; ++ns) {
      int n = wn * 64 + ns * 16 + l16;
      v8bf lo = *(const v8bf*)&sB[n * LDS_S + half * 8];
      v8bf hi = *(const v8bf*)&sB[n * LDS_S + 16 + half * 8];
      bfr[ns] = __builtin_shufflevector(lo, hi, 0,1,2,3,4,5,6,7,8,9,10,11,12,13,14,15);
    }
    #pragma unroll
    for (int ms = 0; ms < 2; ++ms)
      #pragma unroll
      for (int ns = 0; ns < 4; ++ns)
        acc[ms][ns] = __builtin_amdgcn_wmma_f32_16x16x32_bf16(
            false, af[ms], false, bfr[ns], (short)0, acc[ms][ns], false, false);
    __syncthreads();
  }
  #pragma unroll
  for (int ms = 0; ms < 2; ++ms) {
    #pragma unroll
    for (int ns = 0; ns < 4; ++ns) {
      int col = n0 + wn * 64 + ns * 16 + l16;
      int rb = m0 + wm * 32 + ms * 16 + half * 8;
      float bv = bias ? bias[col] : 0.f;
      #pragma unroll
      for (int vi = 0; vi < 8; ++vi) {
        int row = rb + vi;
        if (row < M) {
          float v = acc[ms][ns][vi] + bv;
          if (accum) v += C[(size_t)row * ldc + col];
          if (relu) v = fmaxf(v, 0.f);
          C[(size_t)row * ldc + col] = v;
        }
      }
    }
  }
}

// ---------------------------------------------------------------------------
// Tiny multi-head attention (B=16, S=3, 12 heads x 256): one wave per (b,q,h)
// ---------------------------------------------------------------------------
__global__ void attn_kernel(const float* __restrict__ qh, const float* __restrict__ kh,
                            const float* __restrict__ vh, float* __restrict__ ao) {
  int gid = blockIdx.x * blockDim.x + threadIdx.x;
  int w = gid >> 5, lane = gid & 31;
  if (w >= BB * 3 * AH) return;
  int h = w % AH, t2 = w / AH;
  int qp = t2 % 3, b = t2 / 3;
  const float* qr = qh + ((size_t)(b * 3 + qp) * AH + h) * HD;
  float sc[3];
  #pragma unroll
  for (int kp = 0; kp < 3; ++kp) {
    const float* kr = kh + ((size_t)(b * 3 + kp) * AH + h) * HD;
    float s = 0.f;
    for (int d = lane; d < HD; d += 32) s += qr[d] * kr[d];
    #pragma unroll
    for (int off = 16; off > 0; off >>= 1) s += __shfl_xor(s, off, 32);
    sc[kp] = s * (1.0f / 16.0f);
  }
  float mx = fmaxf(sc[0], fmaxf(sc[1], sc[2]));
  float e0 = __expf(sc[0] - mx), e1 = __expf(sc[1] - mx), e2 = __expf(sc[2] - mx);
  float inv = 1.f / (e0 + e1 + e2);
  float a0 = e0 * inv, a1 = e1 * inv, a2 = e2 * inv;
  float* orow = ao + ((size_t)(b * 3 + qp) * AH + h) * HD;
  for (int d = lane; d < HD; d += 32) {
    orow[d] = a0 * vh[((size_t)(b * 3 + 0) * AH + h) * HD + d]
            + a1 * vh[((size_t)(b * 3 + 1) * AH + h) * HD + d]
            + a2 * vh[((size_t)(b * 3 + 2) * AH + h) * HD + d];
  }
}

__global__ void concat_hidden(const float* __restrict__ ld, const float* __restrict__ bert,
                              const float* __restrict__ fa, float* __restrict__ hidden) {
  int i = blockIdx.x * blockDim.x + threadIdx.x;
  if (i >= BB * 1792) return;
  int b = i / 1792, c = i - b * 1792;
  float v;
  if (c < 256) v = ld[b * 256 + c];
  else if (c < 1024) v = bert[b * 768 + (c - 256)];
  else v = fa[b * 768 + (c - 1024)];
  hidden[i] = v;
}

__global__ void mlp2_kernel(const float* __restrict__ h, const float* __restrict__ W,
                            const float* __restrict__ b, float* __restrict__ out) {
  int t = threadIdx.x;
  if (t >= 32) return;
  int bb = t >> 1, j = t & 1;
  float s = b[j];
  for (int k = 0; k < 896; ++k) s += h[bb * 896 + k] * W[k * 2 + j];
  out[bb * 2 + j] = s;
}

// ---------------------------------------------------------------------------
extern "C" void kernel_launch(void* const* d_in, const int* in_sizes, int n_in,
                              void* d_out, int out_size, void* d_ws, size_t ws_size,
                              hipStream_t stream) {
  auto cdiv = [](int a, int b) { return (a + b - 1) / b; };
  const float* entities_x = (const float*)d_in[0];
  const float* message_x  = (const float*)d_in[1];
  const float* father_x   = (const float*)d_in[2];
  const int*   e_bel  = (const int*)d_in[4];
  const int*   e_cooc = (const int*)d_in[5];
  const int*   e_ref  = (const int*)d_in[6];
  const float* logits = (const float*)d_in[7];
  const float* bert   = (const float*)d_in[8];
  int Ebel = in_sizes[4] / 2, Ecooc = in_sizes[5] / 2, Eref = in_sizes[6] / 2;
  auto F = [&](int i) { return (const float*)d_in[i]; };
  // params (dict insertion order; 'has' params at 9..16 unused)
  const float *W1_bel = F(17), *as1_bel = F(18), *ad1_bel = F(19), *b1_bel = F(20);
  const float *W1_coo = F(25), *as1_coo = F(26), *ad1_coo = F(27), *b1_coo = F(28);
  const float *W1_ref = F(33), *as1_ref = F(34), *ad1_ref = F(35), *b1_ref = F(36);
  const float *W2_ref = F(37), *as2_ref = F(38), *ad2_ref = F(39), *b2_ref = F(40);
  const float *Wq = F(41), *Wk = F(42), *Wv = F(43), *Wiq = F(44), *Wik = F(45), *Wiv = F(46), *Wo = F(47);
  const float *bq = F(48), *bk = F(49), *bv = F(50), *biq = F(51), *bik = F(52), *biv = F(53), *bo = F(54);
  const float *Wt = F(55), *bt = F(56), *Wld = F(57), *bld = F(58);
  const float *Wm1 = F(59), *bm1 = F(60), *Wm2 = F(61), *bm2 = F(62);

  // ---- workspace carve ----
  char* base = (char*)d_ws;
  size_t off = 0;
  auto alloc = [&](size_t bytes) { void* p = base + off; off = (off + bytes + 255) & ~(size_t)255; return p; };
  const int Kp300 = 320;
  bf16* w1tb_bel = (bf16*)alloc((size_t)3072 * Kp300 * 2);
  bf16* w1tb_coo = (bf16*)alloc((size_t)3072 * Kp300 * 2);
  bf16* w1tb_ref = (bf16*)alloc((size_t)3072 * Kp300 * 2);
  bf16* w2tb_ref = (bf16*)alloc((size_t)768 * 3072 * 2);
  bf16* wqt = (bf16*)alloc((size_t)3072 * 3072 * 2);
  bf16* wkt = (bf16*)alloc((size_t)3072 * 3072 * 2);
  bf16* wvt = (bf16*)alloc((size_t)3072 * 3072 * 2);
  bf16* wiqt = (bf16*)alloc((size_t)3072 * 3072 * 2);
  bf16* wikt = (bf16*)alloc((size_t)3072 * 3072 * 2);
  bf16* wivt = (bf16*)alloc((size_t)3072 * 3072 * 2);
  bf16* wot = (bf16*)alloc((size_t)3072 * 3072 * 2);
  bf16* wttb = (bf16*)alloc((size_t)3 * 256 * 3072 * 2);
  bf16* wldt = (bf16*)alloc((size_t)256 * 768 * 2);
  bf16* wm1t = (bf16*)alloc((size_t)896 * 1792 * 2);
  float* wr_as_bel = (float*)alloc(DIN * H1 * 4);
  float* wr_ad_bel = (float*)alloc(DIN * H1 * 4);
  float* wr_as_coo = (float*)alloc(DIN * H1 * 4);
  float* wr_ad_coo = (float*)alloc(DIN * H1 * 4);
  float* wr_as_ref = (float*)alloc(DIN * H1 * 4);
  float* wr_ad_ref = (float*)alloc(DIN * H1 * 4);
  float* wr_as2 = (float*)alloc(3072 * 4);
  float* wr_ad2 = (float*)alloc(3072 * 4);
  float* as_msg = (float*)alloc((size_t)NM * H1 * 4);
  float* ad_ent_bel = (float*)alloc((size_t)NE * H1 * 4);
  float* as_ent_coo = (float*)alloc((size_t)NE * H1 * 4);
  float* ad_ent_coo = (float*)alloc((size_t)NE * H1 * 4);
  float* as_ent_ref = (float*)alloc((size_t)NE * H1 * 4);
  float* ad_fa_ref = (float*)alloc((size_t)BB * H1 * 4);
  float* as2_ent = (float*)alloc((size_t)NE * 4);
  float* ad2_fa = (float*)alloc((size_t)BB * 4);
  int* offs_bel = (int*)alloc((NE + 1) * 4); int* cur_bel = (int*)alloc(NE * 4); int* eix_bel = (int*)alloc((size_t)Ebel * 4);
  int* offs_coo = (int*)alloc((NE + 1) * 4); int* cur_coo = (int*)alloc(NE * 4); int* eix_coo = (int*)alloc((size_t)Ecooc * 4);
  int* offs_ref = (int*)alloc((BB + 1) * 4); int* cur_ref = (int*)alloc(BB * 4); int* eix_ref = (int*)alloc((size_t)Eref * 4);
  float* m_bel = (float*)alloc((size_t)NE * H1 * 4); float* d_bel = (float*)alloc((size_t)NE * H1 * 4);
  float* m_coo = (float*)alloc((size_t)NE * H1 * 4); float* d_coo = (float*)alloc((size_t)NE * H1 * 4);
  float* m_ref = (float*)alloc((size_t)BB * H1 * 4); float* d_ref = (float*)alloc((size_t)BB * H1 * 4);
  float* m2 = (float*)alloc(BB * 4); float* d2 = (float*)alloc(BB * 4);
  float* al_bel = (float*)alloc((size_t)Ebel * H1 * 4);
  float* al_coo = (float*)alloc((size_t)Ecooc * H1 * 4);
  float* al_ref = (float*)alloc((size_t)Eref * H1 * 4);
  float* al2 = (float*)alloc((size_t)Eref * 4);
  float* Zec = (float*)alloc((size_t)NE * H1 * DIN * 4);     // reused belong->cooc
  float* Zrf = (float*)alloc((size_t)BB * H1 * DIN * 4);
  float* Z2 = (float*)alloc((size_t)BB * 3072 * 4);
  float* xd1_ent = (float*)alloc((size_t)NE * 3072 * 4);
  float* xd1_fa = (float*)alloc((size_t)BB * 3072 * 4);
  float* xd2_fa = (float*)alloc((size_t)BB * 768 * 4);
  float* qb = (float*)alloc((size_t)48 * 3072 * 4); float* kb = (float*)alloc((size_t)48 * 3072 * 4);
  float* vb = (float*)alloc((size_t)48 * 3072 * 4);
  float* qhb = (float*)alloc((size_t)48 * 3072 * 4); float* khb = (float*)alloc((size_t)48 * 3072 * 4);
  float* vhb = (float*)alloc((size_t)48 * 3072 * 4);
  float* aop = (float*)alloc((size_t)48 * 3072 * 4); float* aob = (float*)alloc((size_t)48 * 3072 * 4);
  float* ldcat = (float*)alloc((size_t)BB * 768 * 4);
  float* ldv = (float*)alloc((size_t)BB * 256 * 4);
  float* hiddenb = (float*)alloc((size_t)BB * 1792 * 4);
  float* hid2 = (float*)alloc((size_t)BB * 896 * 4);
  (void)ws_size; (void)n_in; (void)out_size;

  dim3 tb(32, 8);
  auto wconv = [&](const float* W, bf16* T, int K, int N, int Kp) {
    wconv_t<<<dim3(cdiv(N, 32), Kp / 32), tb, 0, stream>>>(W, T, K, N, Kp);
  };
  auto gemm = [&](const float* A, int lda, const bf16* Bt, int ldb, float* Cc, int ldc,
                  const float* bias, int M, int N, int K, int acc, int relu) {
    gemm_bf16<<<dim3(N / 128, cdiv(M, 128)), 256, 0, stream>>>(A, lda, Bt, ldb, Cc, ldc, bias, M, N, K, acc, relu);
  };
  auto buildCSR = [&](const int* dstc, int E, int n, int* offs, int* cur, int* eix) {
    hipMemsetAsync(cur, 0, (size_t)n * 4, stream);
    hist_kernel<<<cdiv(E, 256), 256, 0, stream>>>(dstc, E, cur);
    scan_kernel<<<1, 1024, 0, stream>>>(cur, offs, n);
    copy_i32<<<cdiv(n, 256), 256, 0, stream>>>(offs, cur, n);
    fill_kernel<<<cdiv(E, 256), 256, 0, stream>>>(dstc, E, cur, eix);
  };

  // ---- 0. weight conversions (transpose to [N,Kp] bf16) ----
  wconv(W1_bel, w1tb_bel, DIN, 3072, Kp300);
  wconv(W1_coo, w1tb_coo, DIN, 3072, Kp300);
  wconv(W1_ref, w1tb_ref, DIN, 3072, Kp300);
  wconv(W2_ref, w2tb_ref, 3072, 768, 3072);
  wconv(Wq, wqt, 3072, 3072, 3072); wconv(Wk, wkt, 3072, 3072, 3072); wconv(Wv, wvt, 3072, 3072, 3072);
  wconv(Wiq, wiqt, 3072, 3072, 3072); wconv(Wik, wikt, 3072, 3072, 3072); wconv(Wiv, wivt, 3072, 3072, 3072);
  wconv(Wo, wot, 3072, 3072, 3072);
  for (int s = 0; s < 3; ++s) wconv(Wt + (size_t)s * 3072 * 256, wttb + (size_t)s * 256 * 3072, 3072, 256, 3072);
  wconv(Wld, wldt, 768, 256, 768);
  wconv(Wm1, wm1t, 1792, 896, 1792);

  // ---- 1. reduced attention weights W~ = W @ att ----
  redw_kernel<<<cdiv(DIN * H1, 256), 256, 0, stream>>>(W1_bel, as1_bel, wr_as_bel, DIN, H1, C_);
  redw_kernel<<<cdiv(DIN * H1, 256), 256, 0, stream>>>(W1_bel, ad1_bel, wr_ad_bel, DIN, H1, C_);
  redw_kernel<<<cdiv(DIN * H1, 256), 256, 0, stream>>>(W1_coo, as1_coo, wr_as_coo, DIN, H1, C_);
  redw_kernel<<<cdiv(DIN * H1, 256), 256, 0, stream>>>(W1_coo, ad1_coo, wr_ad_coo, DIN, H1, C_);
  redw_kernel<<<cdiv(DIN * H1, 256), 256, 0, stream>>>(W1_ref, as1_ref, wr_as_ref, DIN, H1, C_);
  redw_kernel<<<cdiv(DIN * H1, 256), 256, 0, stream>>>(W1_ref, ad1_ref, wr_ad_ref, DIN, H1, C_);
  redw_kernel<<<cdiv(3072, 256), 256, 0, stream>>>(W2_ref, as2_ref, wr_as2, 3072, 1, C_);
  redw_kernel<<<cdiv(3072, 256), 256, 0, stream>>>(W2_ref, ad2_ref, wr_ad2, 3072, 1, C_);

  // ---- 2. layer-1 attention logits a = X @ W~ ----
  matvec_h<H1><<<NM, 128, 0, stream>>>(message_x, wr_as_bel, as_msg, DIN);
  matvec_h<H1><<<NE, 128, 0, stream>>>(entities_x, wr_ad_bel, ad_ent_bel, DIN);
  matvec_h<H1><<<NE, 128, 0, stream>>>(entities_x, wr_as_coo, as_ent_coo, DIN);
  matvec_h<H1><<<NE, 128, 0, stream>>>(entities_x, wr_ad_coo, ad_ent_coo, DIN);
  matvec_h<H1><<<NE, 128, 0, stream>>>(entities_x, wr_as_ref, as_ent_ref, DIN);
  matvec_h<H1><<<BB, 128, 0, stream>>>(father_x, wr_ad_ref, ad_fa_ref, DIN);

  // ---- 3. CSR by dst ----
  buildCSR(e_bel + Ebel, Ebel, NE, offs_bel, cur_bel, eix_bel);
  buildCSR(e_cooc + Ecooc, Ecooc, NE, offs_coo, cur_coo, eix_coo);
  buildCSR(e_ref + Eref, Eref, BB, offs_ref, cur_ref, eix_ref);

  // ---- 4. layer-1 belong (msg -> entities) ----
  gat_stats<H1><<<NE, 128, 0, stream>>>(offs_bel, eix_bel, e_bel, as_msg, ad_ent_bel, m_bel, d_bel);
  gat_alpha<H1><<<cdiv(Ebel * H1, 256), 256, 0, stream>>>(e_bel, e_bel + Ebel, Ebel, as_msg, ad_ent_bel, m_bel, d_bel, al_bel);
  gat_gather<H1, 3><<<NE, 128, 0, stream>>>(offs_bel, eix_bel, e_bel, al_bel, message_x, DIN, Zec);
  for (int h = 0; h < H1; ++h)
    gemm(Zec + h * DIN, H1 * DIN, w1tb_bel + (size_t)h * C_ * Kp300, Kp300,
         xd1_ent + h * C_, HC1, b1_bel + h * C_, NE, C_, DIN, 0, 0);

  // ---- 5. layer-1 cooc (ent -> entities), accumulate ----
  gat_stats<H1><<<NE, 128, 0, stream>>>(offs_coo, eix_coo, e_cooc, as_ent_coo, ad_ent_coo, m_coo, d_coo);
  gat_alpha<H1><<<cdiv(Ecooc * H1, 256), 256, 0, stream>>>(e_cooc, e_cooc + Ecooc, Ecooc, as_ent_coo, ad_ent_coo, m_coo, d_coo, al_coo);
  gat_gather<H1, 3><<<NE, 128, 0, stream>>>(offs_coo, eix_coo, e_cooc, al_coo, entities_x, DIN, Zec);
  for (int h = 0; h < H1; ++h)
    gemm(Zec + h * DIN, H1 * DIN, w1tb_coo + (size_t)h * C_ * Kp300, Kp300,
         xd1_ent + h * C_, HC1, b1_coo + h * C_, NE, C_, DIN, 1, 0);

  // ---- 6. layer-1 refer (ent -> father) ----
  gat_stats<H1><<<BB, 128, 0, stream>>>(offs_ref, eix_ref, e_ref, as_ent_ref, ad_fa_ref, m_ref, d_ref);
  gat_alpha<H1><<<cdiv(Eref * H1, 256), 256, 0, stream>>>(e_ref, e_ref + Eref, Eref, as_ent_ref, ad_fa_ref, m_ref, d_ref, al_ref);
  hipMemsetAsync(Zrf, 0, (size_t)BB * H1 * DIN * 4, stream);
  gat_gather_split<H1, 3><<<dim3(BB, 64), 128, 0, stream>>>(offs_ref, eix_ref, e_ref, al_ref, entities_x, DIN, Zrf);
  for (int h = 0; h < H1; ++h)
    gemm(Zrf + h * DIN, H1 * DIN, w1tb_ref + (size_t)h * C_ * Kp300, Kp300,
         xd1_fa + h * C_, HC1, b1_ref + h * C_, BB, C_, DIN, 0, 0);

  // ---- 7. layer-2 logits ----
  matvec_h<1><<<NE, 128, 0, stream>>>(xd1_ent, wr_as2, as2_ent, HC1);
  matvec_h<1><<<BB, 128, 0, stream>>>(xd1_fa, wr_ad2, ad2_fa, HC1);

  // ---- 8. layer-2 refer (only relation feeding the output) ----
  gat_stats<1><<<BB, 128, 0, stream>>>(offs_ref, eix_ref, e_ref, as2_ent, ad2_fa, m2, d2);
  gat_alpha<1><<<cdiv(Eref, 256), 256, 0, stream>>>(e_ref, e_ref + Eref, Eref, as2_ent, ad2_fa, m2, d2, al2);
  hipMemsetAsync(Z2, 0, (size_t)BB * HC1 * 4, stream);
  gat_gather_split<1, 24><<<dim3(BB, 64), 128, 0, stream>>>(offs_ref, eix_ref, e_ref, al2, xd1_ent, HC1, Z2);
  gemm(Z2, HC1, w2tb_ref, 3072, xd2_fa, 768, b2_ref, BB, 768, HC1, 0, 0);

  // ---- 9. attention branch ----
  gemm(logits, EDIM, wqt, 3072, qb, EDIM, bq, 48, EDIM, EDIM, 0, 0);
  gemm(logits, EDIM, wkt, 3072, kb, EDIM, bk, 48, EDIM, EDIM, 0, 0);
  gemm(logits, EDIM, wvt, 3072, vb, EDIM, bv, 48, EDIM, EDIM, 0, 0);
  gemm(qb, EDIM, wiqt, 3072, qhb, EDIM, biq, 48, EDIM, EDIM, 0, 0);
  gemm(kb, EDIM, wikt, 3072, khb, EDIM, bik, 48, EDIM, EDIM, 0, 0);
  gemm(vb, EDIM, wivt, 3072, vhb, EDIM, biv, 48, EDIM, EDIM, 0, 0);
  attn_kernel<<<cdiv(BB * 3 * AH * 32, 256), 256, 0, stream>>>(qhb, khb, vhb, aop);
  gemm(aop, EDIM, wot, 3072, aob, EDIM, bo, 48, EDIM, EDIM, 0, 0);
  for (int s = 0; s < 3; ++s)
    gemm(aob + s * EDIM, 3 * EDIM, wttb + (size_t)s * 256 * 3072, 3072,
         ldcat + s * 256, 768, bt + s * 256, BB, 256, EDIM, 0, 0);
  gemm(ldcat, 768, wldt, 768, ldv, 256, bld, BB, 256, 768, 0, 0);
  concat_hidden<<<cdiv(BB * 1792, 256), 256, 0, stream>>>(ldv, bert, xd2_fa, hiddenb);
  gemm(hiddenb, 1792, wm1t, 1792, hid2, 896, bm1, BB, 896, 1792, 0, 1);
  mlp2_kernel<<<1, 32, 0, stream>>>(hid2, Wm2, bm2, (float*)d_out);
}
